// LLaDA2SparseMoeBlock_53309134078045
// MI455X (gfx1250) — compile-verified, matrix operand order checked by reading, not run on hardware
//
#include <hip/hip_runtime.h>
#include <hip/hip_bf16.h>

// ---------------- problem constants (match reference) ----------------
constexpr int cE = 32;    // experts
constexpr int cK = 4;     // top-k
constexpr int cH = 1024;  // hidden
constexpr int cI = 512;   // intermediate
constexpr int cT = 4096;  // tokens (B*S)
constexpr int cC = 1024;  // capacity per expert

using u16 = unsigned short;

// ---------------- WMMA plumbing (gfx1250, wave32) ----------------
typedef __attribute__((ext_vector_type(16))) __bf16 bf16x16;
typedef __attribute__((ext_vector_type(8)))  float  f32x8;

union FragU { uint4 q[2]; bf16x16 v; };

__device__ __forceinline__ bf16x16 load_frag(const u16* p) {
  // p points at K-base for this lane; frag = {p[0..7], p[16..23]} (16-bit A/B layout)
  FragU f;
  f.q[0] = *reinterpret_cast<const uint4*>(p);
  f.q[1] = *reinterpret_cast<const uint4*>(p + 16);
  return f.v;
}

__device__ __forceinline__ f32x8 wmma_bf16(bf16x16 a, bf16x16 b, f32x8 c) {
  return __builtin_amdgcn_wmma_f32_16x16x32_bf16(false, a, false, b, (short)0, c, false, false);
}

__device__ __forceinline__ u16 f2bf(float x) {
  union { float f; unsigned u; } v; v.f = x;
  unsigned r = v.u + 0x7FFFu + ((v.u >> 16) & 1u);  // round-to-nearest-even
  return (u16)(r >> 16);
}

__device__ __forceinline__ float silu_mul(float g, float u) {
  return (g / (1.0f + __expf(-g))) * u;
}

// ---------------- conversion kernels ----------------
__global__ void convert_bf16_kernel(const float* __restrict__ src, u16* __restrict__ dst, size_t n4) {
  // processes 4 floats per thread; n4 = n/4
  size_t i = (size_t)blockIdx.x * blockDim.x + threadIdx.x;
  size_t stride = (size_t)gridDim.x * blockDim.x;
  const float4* s4 = reinterpret_cast<const float4*>(src);
  uint2* d2 = reinterpret_cast<uint2*>(dst);
  for (; i < n4; i += stride) {
    float4 v = s4[i];
    uint2 o;
    o.x = (unsigned)f2bf(v.x) | ((unsigned)f2bf(v.y) << 16);
    o.y = (unsigned)f2bf(v.z) | ((unsigned)f2bf(v.w) << 16);
    d2[i] = o;
  }
}

__global__ void zero_u16_kernel(u16* __restrict__ p, int n) {
  int i = blockIdx.x * blockDim.x + threadIdx.x;
  if (i < n) p[i] = 0;
}

// src [Z][R][Cc] f32 -> dst [Z][Cc][R] bf16  (tiled transpose)
__global__ __launch_bounds__(256) void transpose_f32_bf16_kernel(
    const float* __restrict__ src, u16* __restrict__ dst, int R, int Cc) {
  __shared__ float tile[32][33];
  int z = blockIdx.z;
  src += (size_t)z * R * Cc;
  dst += (size_t)z * R * Cc;
  int c0 = blockIdx.x * 32, r0 = blockIdx.y * 32;
  int tx = threadIdx.x, ty = threadIdx.y;  // (32, 8)
  for (int i = 0; i < 32; i += 8)
    tile[ty + i][tx] = src[(size_t)(r0 + ty + i) * Cc + (c0 + tx)];
  __syncthreads();
  for (int i = 0; i < 32; i += 8)
    dst[(size_t)(c0 + ty + i) * R + (r0 + tx)] = f2bf(tile[tx][ty + i]);
}

// ---------------- router: logits -> softmax -> top4 -> renorm ----------------
__global__ __launch_bounds__(128) void router_kernel(
    const float* __restrict__ x, const float* __restrict__ gw,
    int* __restrict__ tidx, float* __restrict__ tw) {
  __shared__ float xs[cH];
  __shared__ float part[cE][4];
  int t = blockIdx.x;
  int tid = threadIdx.x;

  const float4* x4 = reinterpret_cast<const float4*>(x + (size_t)t * cH);
  float4* xs4 = reinterpret_cast<float4*>(xs);
  for (int i = tid; i < cH / 4; i += 128) xs4[i] = x4[i];
  __syncthreads();

  int e = tid & 31, c = tid >> 5;  // 4 threads per expert
  const float* gwe = gw + (size_t)e * cH + c * (cH / 4);
  const float* xc = xs + c * (cH / 4);
  float s = 0.f;
  for (int k = 0; k < cH / 4; ++k) s += xc[k] * gwe[k];
  part[e][c] = s;
  __syncthreads();

  if (tid < 32) {  // one full wave32: shuffle softmax + top-4
    float l = part[tid][0] + part[tid][1] + part[tid][2] + part[tid][3];
    float mx = l;
    for (int off = 16; off; off >>= 1) mx = fmaxf(mx, __shfl_xor(mx, off, 32));
    float ex = __expf(l - mx);
    float sm = ex;
    for (int off = 16; off; off >>= 1) sm += __shfl_xor(sm, off, 32);
    float myp = ex / sm;

    float wv[cK]; int wi[cK];
    for (int k = 0; k < cK; ++k) {
      float bv = myp; int bi = tid;
      for (int off = 16; off; off >>= 1) {
        float ov = __shfl_xor(bv, off, 32);
        int   oi = __shfl_xor(bi, off, 32);
        if (ov > bv || (ov == bv && oi < bi)) { bv = ov; bi = oi; }
      }
      wv[k] = bv; wi[k] = bi;
      if (tid == bi) myp = -1.0f;  // remove selected
    }
    if (tid == 0) {
      float ssum = wv[0] + wv[1] + wv[2] + wv[3];
      for (int k = 0; k < cK; ++k) {
        tw[(size_t)t * cK + k] = wv[k] / ssum;  // ROUTED_SCALE == 1.0
        tidx[(size_t)t * cK + k] = wi[k];
      }
    }
  }
}

// ---------------- dispatch ----------------
__global__ void zero_counts_kernel(int* counts) {
  if (threadIdx.x < cE) counts[threadIdx.x] = 0;
}

__global__ void dispatch_kernel(const int* __restrict__ tidx, const float* __restrict__ tw,
                                int* __restrict__ counts, int* __restrict__ token_of,
                                float* __restrict__ gate_of) {
  int i = blockIdx.x * blockDim.x + threadIdx.x;
  if (i >= cT * cK) return;
  int t = i >> 2;  // cK == 4
  int e = tidx[i];
  float w = tw[i];
  int pos = atomicAdd(&counts[e], 1);
  if (pos < cC) {
    token_of[e * cC + pos] = t;
    gate_of[e * cC + pos] = w;
  }
}

// ---------------- GEMM1: act = silu(X@Wg) * (X@Wu), bf16 WMMA ----------------
// WT is the TRANSPOSED gate_up weight: [2I rows, H cols] per expert.
// Per wave: 64(M) x 32(N-act), gate+up: 8 fragment loads -> 16 WMMA (1:1 vmem:wmma).
// Block: 8 waves (2M x 4N) = 128M x 128N-act. K-loop is branch-free: invalid
// rows point at a zeroed dummy row (Zrow), no EXEC divergence inside the loop.
__global__ __launch_bounds__(256) void gemm1_kernel(
    const u16* __restrict__ X, const u16* __restrict__ WT, u16* __restrict__ ACT,
    const int* __restrict__ token_of, const int* __restrict__ counts,
    const u16* __restrict__ Zrow, int Mcap) {
  const int e = blockIdx.z;
  const u16* WTe = WT + (size_t)e * (2 * cI) * cH;
  u16* ACTe = ACT + (size_t)e * Mcap * cI;
  const int* toke = token_of ? token_of + (size_t)e * Mcap : nullptr;
  int cnt = Mcap;
  if (counts) { cnt = counts[e]; if (cnt > Mcap) cnt = Mcap; }
  int mblock = blockIdx.y * 128;
  if (mblock >= cnt) return;

  int wave = threadIdx.x >> 5, lane = threadIdx.x & 31;
  int l16 = lane & 15, hi = lane >> 4;
  int koff = hi ? 8 : 0;
  int wm = wave >> 2, wn = wave & 3;
  int m0 = mblock + wm * 64;
  int n0 = blockIdx.x * 128 + wn * 32;  // act-column tile base (32 wide)

  // A row pointers (4 m-subtiles), per-lane; invalid rows -> zero row
  const u16* ra[4];
#pragma unroll
  for (int mi = 0; mi < 4; ++mi) {
    int mr = m0 + mi * 16 + l16;
    if (toke) {
      int s = mr < cnt ? mr : 0;  // cnt >= 1 here (early-exit above)
      ra[mi] = (mr < cnt) ? X + (size_t)toke[s] * cH : Zrow;
    } else {
      ra[mi] = X + (size_t)mr * cH;
    }
  }
  const u16* rbg[2];
  const u16* rbu[2];
#pragma unroll
  for (int ni = 0; ni < 2; ++ni) {
    rbg[ni] = WTe + (size_t)(n0 + ni * 16 + l16) * cH;       // gate cols
    rbu[ni] = WTe + (size_t)(cI + n0 + ni * 16 + l16) * cH;  // up cols
  }

  f32x8 z8 = {0, 0, 0, 0, 0, 0, 0, 0};
  f32x8 cg[4][2], cu[4][2];
#pragma unroll
  for (int mi = 0; mi < 4; ++mi)
#pragma unroll
    for (int ni = 0; ni < 2; ++ni) { cg[mi][ni] = z8; cu[mi][ni] = z8; }

  for (int kk = 0; kk < cH; kk += 32) {
    int k0 = kk + koff;
    bf16x16 a[4], bg[2], bu[2];
#pragma unroll
    for (int mi = 0; mi < 4; ++mi) a[mi] = load_frag(ra[mi] + k0);
#pragma unroll
    for (int ni = 0; ni < 2; ++ni) {
      bg[ni] = load_frag(rbg[ni] + k0);
      bu[ni] = load_frag(rbu[ni] + k0);
    }
#pragma unroll
    for (int mi = 0; mi < 4; ++mi)
#pragma unroll
      for (int ni = 0; ni < 2; ++ni) {
        cg[mi][ni] = wmma_bf16(a[mi], bg[ni], cg[mi][ni]);
        cu[mi][ni] = wmma_bf16(a[mi], bu[ni], cu[mi][ni]);
      }
  }

#pragma unroll
  for (int mi = 0; mi < 4; ++mi) {
#pragma unroll
    for (int r = 0; r < 8; ++r) {
      int mrow = m0 + mi * 16 + r + hi * 8;
      u16* arow = ACTe + (size_t)mrow * cI;
      arow[n0 + l16]      = f2bf(silu_mul(cg[mi][0][r], cu[mi][0][r]));
      arow[n0 + 16 + l16] = f2bf(silu_mul(cg[mi][1][r], cu[mi][1][r]));
    }
  }
}

// ---------------- GEMM2: out = act @ Wd (WT = transposed w_down: [H rows, I cols]) ----
// Per wave: 64(M) x 64(N): 8 fragment loads -> 16 WMMA, 16 accumulators.
// Block: 8 waves (2M x 4N) = 128M x 256N.
__global__ __launch_bounds__(256) void gemm2_kernel(
    const u16* __restrict__ ACT, const u16* __restrict__ WT, float* __restrict__ OUT,
    const int* __restrict__ token_of, const float* __restrict__ gate_of,
    const int* __restrict__ counts, int Mcap) {
  const int e = blockIdx.z;
  const u16* ACTe = ACT + (size_t)e * Mcap * cI;
  const u16* WTe = WT + (size_t)e * cH * cI;
  const int* toke = token_of ? token_of + (size_t)e * Mcap : nullptr;
  const float* gte = gate_of ? gate_of + (size_t)e * Mcap : nullptr;
  int cnt = Mcap;
  if (counts) { cnt = counts[e]; if (cnt > Mcap) cnt = Mcap; }
  int mblock = blockIdx.y * 128;
  if (mblock >= cnt) return;

  int wave = threadIdx.x >> 5, lane = threadIdx.x & 31;
  int l16 = lane & 15, hi = lane >> 4;
  int koff = hi ? 8 : 0;
  int wm = wave >> 2, wn = wave & 3;
  int m0 = mblock + wm * 64;
  int n0 = blockIdx.x * 256 + wn * 64;

  const u16* ra[4];
  const u16* rb[4];
#pragma unroll
  for (int i = 0; i < 4; ++i) {
    ra[i] = ACTe + (size_t)(m0 + i * 16 + l16) * cI;
    rb[i] = WTe + (size_t)(n0 + i * 16 + l16) * cI;
  }

  f32x8 z8 = {0, 0, 0, 0, 0, 0, 0, 0};
  f32x8 c[4][4];
#pragma unroll
  for (int mi = 0; mi < 4; ++mi)
#pragma unroll
    for (int ni = 0; ni < 4; ++ni) c[mi][ni] = z8;

  for (int kk = 0; kk < cI; kk += 32) {
    int k0 = kk + koff;
    bf16x16 a[4], b[4];
#pragma unroll
    for (int i = 0; i < 4; ++i) a[i] = load_frag(ra[i] + k0);
#pragma unroll
    for (int i = 0; i < 4; ++i) b[i] = load_frag(rb[i] + k0);
#pragma unroll
    for (int mi = 0; mi < 4; ++mi)
#pragma unroll
      for (int ni = 0; ni < 4; ++ni)
        c[mi][ni] = wmma_bf16(a[mi], b[ni], c[mi][ni]);
  }

  if (toke) {  // routed experts: gated scatter-add into out
#pragma unroll
    for (int mi = 0; mi < 4; ++mi) {
#pragma unroll
      for (int r = 0; r < 8; ++r) {
        int mr = m0 + mi * 16 + r + hi * 8;
        if (mr < cnt) {
          int tt = toke[mr];
          float gv = gte[mr];
          float* o = OUT + (size_t)tt * cH;
          atomicAdd(&o[n0 + l16],      c[mi][0][r] * gv);
          atomicAdd(&o[n0 + 16 + l16], c[mi][1][r] * gv);
          atomicAdd(&o[n0 + 32 + l16], c[mi][2][r] * gv);
          atomicAdd(&o[n0 + 48 + l16], c[mi][3][r] * gv);
        }
      }
    }
  } else {  // shared expert: plain store (initializes d_out)
#pragma unroll
    for (int mi = 0; mi < 4; ++mi) {
#pragma unroll
      for (int r = 0; r < 8; ++r) {
        int mr = m0 + mi * 16 + r + hi * 8;
        float* o = OUT + (size_t)mr * cH;
        o[n0 + l16]      = c[mi][0][r];
        o[n0 + 16 + l16] = c[mi][1][r];
        o[n0 + 32 + l16] = c[mi][2][r];
        o[n0 + 48 + l16] = c[mi][3][r];
      }
    }
  }
}

// ---------------- launcher ----------------
extern "C" void kernel_launch(void* const* d_in, const int* in_sizes, int n_in,
                              void* d_out, int out_size, void* d_ws, size_t ws_size,
                              hipStream_t stream) {
  (void)in_sizes; (void)n_in; (void)out_size; (void)ws_size;
  const float* x          = (const float*)d_in[0];  // [B,S,H]
  const float* gate_w     = (const float*)d_in[1];  // [E,H]
  const float* w_gate_up  = (const float*)d_in[2];  // [E,H,2I]
  const float* w_down     = (const float*)d_in[3];  // [E,I,H]
  const float* s_gu       = (const float*)d_in[4];  // [H,2I]
  const float* s_d        = (const float*)d_in[5];  // [I,H]
  float* out = (float*)d_out;

  char* ws = (char*)d_ws;
  size_t off = 0;
  auto alloc = [&](size_t bytes) -> char* {
    char* p = ws + off;
    off += (bytes + 255) & ~(size_t)255;
    return p;
  };
  u16*   xb      = (u16*)alloc((size_t)cT * cH * 2);
  u16*   wguT    = (u16*)alloc((size_t)cE * 2 * cI * cH * 2);
  u16*   wdT     = (u16*)alloc((size_t)cE * cH * cI * 2);
  u16*   sguT    = (u16*)alloc((size_t)2 * cI * cH * 2);
  u16*   sdT     = (u16*)alloc((size_t)cH * cI * 2);
  u16*   act_s   = (u16*)alloc((size_t)cT * cI * 2);
  u16*   act_e   = (u16*)alloc((size_t)cE * cC * cI * 2);
  u16*   zrow    = (u16*)alloc((size_t)cH * 2);
  int*   tidx    = (int*)alloc((size_t)cT * cK * 4);
  float* tw      = (float*)alloc((size_t)cT * cK * 4);
  int*   counts  = (int*)alloc((size_t)cE * 4);
  int*   token_of= (int*)alloc((size_t)cE * cC * 4);
  float* gate_of = (float*)alloc((size_t)cE * cC * 4);

  // 1) precision conversion + weight transposes (one pass; bf16 weights are L2-resident)
  convert_bf16_kernel<<<512, 256, 0, stream>>>(x, xb, (size_t)cT * cH / 4);
  zero_u16_kernel<<<(cH + 255) / 256, 256, 0, stream>>>(zrow, cH);
  dim3 tb(32, 8);
  transpose_f32_bf16_kernel<<<dim3(2 * cI / 32, cH / 32, cE), tb, 0, stream>>>(w_gate_up, wguT, cH, 2 * cI);
  transpose_f32_bf16_kernel<<<dim3(cH / 32, cI / 32, cE), tb, 0, stream>>>(w_down, wdT, cI, cH);
  transpose_f32_bf16_kernel<<<dim3(2 * cI / 32, cH / 32, 1), tb, 0, stream>>>(s_gu, sguT, cH, 2 * cI);
  transpose_f32_bf16_kernel<<<dim3(cH / 32, cI / 32, 1), tb, 0, stream>>>(s_d, sdT, cI, cH);

  // 2) router (fp32, exact softmax/top-4) + dispatch
  router_kernel<<<cT, 128, 0, stream>>>(x, gate_w, tidx, tw);
  zero_counts_kernel<<<1, 32, 0, stream>>>(counts);
  dispatch_kernel<<<(cT * cK + 255) / 256, 256, 0, stream>>>(tidx, tw, counts, token_of, gate_of);

  // 3) shared expert (writes every element of d_out first)
  gemm1_kernel<<<dim3(cI / 128, cT / 128, 1), 256, 0, stream>>>(xb, sguT, act_s, nullptr, nullptr, zrow, cT);
  gemm2_kernel<<<dim3(cH / 256, cT / 128, 1), 256, 0, stream>>>(act_s, sdT, out, nullptr, nullptr, nullptr, cT);

  // 4) routed experts (gather rows by token id; gated atomic-add combine)
  gemm1_kernel<<<dim3(cI / 128, cC / 128, cE), 256, 0, stream>>>(xb, wguT, act_e, token_of, counts, zrow, cC);
  gemm2_kernel<<<dim3(cH / 256, cC / 128, cE), 256, 0, stream>>>(act_e, wdT, out, token_of, gate_of, counts, cC);
}
